// DeepseekV3MoEExperts_12223476924456
// MI455X (gfx1250) — matrix operand order repaired and checked by reading, not compiled, not run on hardware
//
#include <hip/hip_runtime.h>
#include <hip/hip_bf16.h>

// ---------------------------------------------------------------------------
// DeepseekV3 MoE experts (all 16 experts over the full token set), MI455X.
//   g = x @ Wg^T ; u = x @ Wu^T ; h = silu(g)*u ; out = h @ Wd^T
// fp32 in/out; fp16 V_WMMA_F32_16X16X32_F16 with f32 accumulation.
// Activations pre-converted to fp16 once; weights converted in-register
// while streaming (384 MB read once -> ~17 us HBM floor at 23.3 TB/s).
// ---------------------------------------------------------------------------

typedef __attribute__((ext_vector_type(16))) _Float16 v16h;
typedef __attribute__((ext_vector_type(8)))  _Float16 v8h;
typedef __attribute__((ext_vector_type(8)))  float    v8f;
typedef __attribute__((ext_vector_type(4)))  float    fv4;

constexpr int E_ = 16;
constexpr int T_ = 256;
constexpr int H_ = 2048;
constexpr int I_ = 1024;

// B-fragment (32x16, 16-bit): lane l holds column l%16,
// elems 0..7 = K[8*(l/16)+j], elems 8..15 = K[16+8*(l/16)+j].  p points at W-row + kb + 8*(l/16).
__device__ __forceinline__ v16h frag_b_f32(const float* __restrict__ p) {
  fv4 a0 = *(const fv4*)(p + 0);
  fv4 a1 = *(const fv4*)(p + 4);
  fv4 a2 = *(const fv4*)(p + 16);
  fv4 a3 = *(const fv4*)(p + 20);
  v16h r;
#pragma unroll
  for (int j = 0; j < 4; ++j) {
    r[j]      = (_Float16)a0[j];
    r[j + 4]  = (_Float16)a1[j];
    r[j + 8]  = (_Float16)a2[j];
    r[j + 12] = (_Float16)a3[j];
  }
  return r;
}

// A-fragment (16x32, 16-bit) from fp16 memory: lane l holds row l%16,
// elems 0..7 = K[ksel..ksel+7], elems 8..15 = K[16+ksel..16+ksel+7],
// ksel = 8*(l/16).  p points at row start + kb.
__device__ __forceinline__ v16h frag_a_f16(const _Float16* __restrict__ p, int ksel) {
  v8h lo = *(const v8h*)(p + ksel);
  v8h hi = *(const v8h*)(p + ksel + 16);
  v16h r;
#pragma unroll
  for (int j = 0; j < 8; ++j) {
    r[j]     = lo[j];
    r[j + 8] = hi[j];
  }
  return r;
}

__device__ __forceinline__ v8f wmma_f16(v16h a, v16h b, v8f c) {
  return __builtin_amdgcn_wmma_f32_16x16x32_f16(
      /*neg_a=*/false, a, /*neg_b=*/false, b,
      /*c_mod=*/(short)0, c, /*reuse_a=*/false, /*reuse_b=*/false);
}

// ---------------------------------------------------------------------------
// Kernel 0: convert x [T,H] fp32 -> fp16 once (2 MB read, 1 MB write).
// 8 elements per thread; grid = T*H/(256*8) = 256 blocks.
// ---------------------------------------------------------------------------
__global__ __launch_bounds__(256) void cvt_x_kernel(
    const float* __restrict__ x, _Float16* __restrict__ xh) {
  const int base = (blockIdx.x * 256 + threadIdx.x) * 8;
  fv4 a0 = *(const fv4*)(x + base);
  fv4 a1 = *(const fv4*)(x + base + 4);
  v8h r;
#pragma unroll
  for (int j = 0; j < 4; ++j) {
    r[j]     = (_Float16)a0[j];
    r[j + 4] = (_Float16)a1[j];
  }
  *(v8h*)(xh + base) = r;
}

// ---------------------------------------------------------------------------
// Kernel 1: h[e,t,i] = silu(x@Wg^T) * (x@Wu^T), stored fp16 [E,T,I].
// Grid: E * (I/32) blocks, 256 threads (8 waves).
// Wave w: M-block = (w%4)*64 (4 subtiles of 16 rows), N-tile = i0 + (w/4)*16.
// Weight frags reused 4x in-register across M subtiles; cross-wave reuse
// (4x along T, both N-groups) served by the 192 MB L2.
// ---------------------------------------------------------------------------
__global__ __launch_bounds__(256) void moe_gateup_kernel(
    const _Float16* __restrict__ xh, const float* __restrict__ wg,
    const float* __restrict__ wu, _Float16* __restrict__ hws) {
  constexpr int NBLK = I_ / 32;
  const int e  = blockIdx.x / NBLK;
  const int i0 = (blockIdx.x % NBLK) * 32;

  const int wave = threadIdx.x >> 5;
  const int lane = threadIdx.x & 31;
  const int m0   = (wave & 3) * 64;
  const int n0   = i0 + (wave >> 2) * 16;
  const int r    = lane & 15;
  const int asel = (lane >> 4) * 8;
  const int bsel = (lane >> 4) * 8;

  const _Float16* xrow[4];
#pragma unroll
  for (int ms = 0; ms < 4; ++ms)
    xrow[ms] = xh + (size_t)(m0 + ms * 16 + r) * H_;
  const float* grow = wg + ((size_t)e * I_ + n0 + r) * H_;
  const float* urow = wu + ((size_t)e * I_ + n0 + r) * H_;

  const v8f zero = {0.f, 0.f, 0.f, 0.f, 0.f, 0.f, 0.f, 0.f};
  v8f accg[4], accu[4];
#pragma unroll
  for (int ms = 0; ms < 4; ++ms) { accg[ms] = zero; accu[ms] = zero; }

#pragma unroll 2
  for (int kb = 0; kb < H_; kb += 32) {
    // prefetch the next chunk of the streaming weight rows (global_prefetch_b8)
    __builtin_prefetch(grow + kb + 64, 0, 1);
    __builtin_prefetch(urow + kb + 64, 0, 1);
    v16h bg = frag_b_f32(grow + kb + bsel);
    v16h bu = frag_b_f32(urow + kb + bsel);
#pragma unroll
    for (int ms = 0; ms < 4; ++ms) {
      v16h a = frag_a_f16(xrow[ms] + kb, asel);
      accg[ms] = wmma_f16(a, bg, accg[ms]);
      accu[ms] = wmma_f16(a, bu, accu[ms]);
    }
  }

  // Epilogue: silu(g)*u, store fp16.  D layout: lane holds N=lane%16,
  // M = 8*(lane/16) + v within each 16-row subtile.
  const int mbase = m0 + 8 * (lane >> 4);
#pragma unroll
  for (int ms = 0; ms < 4; ++ms) {
#pragma unroll
    for (int v = 0; v < 8; ++v) {
      float g  = accg[ms][v];
      float s  = g / (1.0f + __expf(-g));
      float hv = s * accu[ms][v];
      int row  = mbase + ms * 16 + v;
      hws[((size_t)e * T_ + row) * I_ + n0 + r] = (_Float16)hv;
    }
  }
}

// ---------------------------------------------------------------------------
// Kernel 2: out[e,t,hh] = h[e] @ Wd[e]^T   (K = I = 1024), fp32 out.
// Grid: E * (H/32) blocks, 256 threads (8 waves).
// ---------------------------------------------------------------------------
__global__ __launch_bounds__(256) void moe_down_kernel(
    const _Float16* __restrict__ hws, const float* __restrict__ wd,
    float* __restrict__ out) {
  constexpr int NBLK = H_ / 32;
  const int e  = blockIdx.x / NBLK;
  const int h0 = (blockIdx.x % NBLK) * 32;

  const int wave = threadIdx.x >> 5;
  const int lane = threadIdx.x & 31;
  const int m0   = (wave & 3) * 64;
  const int n0   = h0 + (wave >> 2) * 16;
  const int r    = lane & 15;
  const int asel = (lane >> 4) * 8;
  const int bsel = (lane >> 4) * 8;

  const _Float16* arow[4];
#pragma unroll
  for (int ms = 0; ms < 4; ++ms)
    arow[ms] = hws + ((size_t)e * T_ + m0 + ms * 16 + r) * I_;
  const float* brow = wd + ((size_t)e * H_ + n0 + r) * I_;

  const v8f zero = {0.f, 0.f, 0.f, 0.f, 0.f, 0.f, 0.f, 0.f};
  v8f acc[4];
#pragma unroll
  for (int ms = 0; ms < 4; ++ms) acc[ms] = zero;

#pragma unroll 2
  for (int kb = 0; kb < I_; kb += 32) {
    __builtin_prefetch(brow + kb + 64, 0, 1);
    v16h b = frag_b_f32(brow + kb + bsel);
#pragma unroll
    for (int ms = 0; ms < 4; ++ms) {
      v16h a = frag_a_f16(arow[ms] + kb, asel);
      acc[ms] = wmma_f16(a, b, acc[ms]);
    }
  }

  const int mbase = m0 + 8 * (lane >> 4);
#pragma unroll
  for (int ms = 0; ms < 4; ++ms) {
#pragma unroll
    for (int v = 0; v < 8; ++v) {
      int row = mbase + ms * 16 + v;
      out[((size_t)e * T_ + row) * H_ + n0 + r] = acc[ms][v];
    }
  }
}

// ---------------------------------------------------------------------------
extern "C" void kernel_launch(void* const* d_in, const int* in_sizes, int n_in,
                              void* d_out, int out_size, void* d_ws, size_t ws_size,
                              hipStream_t stream) {
  (void)in_sizes; (void)n_in; (void)out_size; (void)ws_size;
  const float* x  = (const float*)d_in[0];  // [T, H]
  const float* wg = (const float*)d_in[1];  // [E, I, H]
  const float* wu = (const float*)d_in[2];  // [E, I, H]
  const float* wd = (const float*)d_in[3];  // [E, H, I]
  float* out      = (float*)d_out;          // [E*T, H]

  _Float16* xh  = (_Float16*)d_ws;          // [T, H]   fp16 (1 MB)
  _Float16* hws = xh + (size_t)T_ * H_;     // [E, T, I] fp16 (8 MB)

  dim3 blk(256);
  cvt_x_kernel<<<dim3((T_ * H_) / (256 * 8)), blk, 0, stream>>>(x, xh);
  moe_gateup_kernel<<<dim3(E_ * (I_ / 32)), blk, 0, stream>>>(xh, wg, wu, hws);
  moe_down_kernel<<<dim3(E_ * (H_ / 32)), blk, 0, stream>>>(hws, wd, out);
}
